// AnomalyGNN_41875931136204
// MI455X (gfx1250) — matrix operand order repaired and checked by reading, not compiled, not run on hardware
//
#include <hip/hip_runtime.h>
#include <hip/hip_bf16.h>

typedef __bf16 bf16;
typedef __attribute__((ext_vector_type(8)))  __bf16 v8bf;
typedef __attribute__((ext_vector_type(16))) __bf16 v16bf;
typedef __attribute__((ext_vector_type(8)))  float  v8f;

__device__ __forceinline__ v16bf cat8(v8bf lo, v8bf hi) {
  return __builtin_shufflevector(lo, hi, 0,1,2,3,4,5,6,7,8,9,10,11,12,13,14,15);
}

// gfx1250 async copy: global -> LDS, 16 bytes per lane, tracked by ASYNCcnt.
__device__ __forceinline__ void async_g2lds_b128(unsigned lds_off, const void* gaddr) {
  asm volatile("global_load_async_to_lds_b128 %0, %1, off"
               :: "v"(lds_off), "v"(gaddr) : "memory");
}
__device__ __forceinline__ void wait_async0() {
  asm volatile("s_wait_asynccnt 0" ::: "memory");
}

// ---------------- degree / norm prep ----------------
__global__ void k_degree(const int* __restrict__ dst, float* __restrict__ deg, int E) {
  int e = blockIdx.x * blockDim.x + threadIdx.x;
  if (e < E)
    __hip_atomic_fetch_add(&deg[dst[e]], 1.0f, __ATOMIC_RELAXED, __HIP_MEMORY_SCOPE_AGENT);
}

__global__ void k_dinv(float* __restrict__ deg, int N) {
  int i = blockIdx.x * blockDim.x + threadIdx.x;
  if (i < N) deg[i] = rsqrtf(deg[i] + 1.0f);   // in-place: deg -> dinv
}

__global__ void k_edge_norm(const int* __restrict__ src, const int* __restrict__ dst,
                            const float* __restrict__ dinv, float* __restrict__ enorm, int E) {
  int e = blockIdx.x * blockDim.x + threadIdx.x;
  if (e < E) enorm[e] = dinv[src[e]] * dinv[dst[e]];
}

// ---------------- conversions ----------------
__global__ void k_f32_to_bf16(const float* __restrict__ in, bf16* __restrict__ out, int n) {
  int i = blockIdx.x * blockDim.x + threadIdx.x;
  if (i < n) out[i] = (bf16)in[i];
}

// W[K,F] (row-major f32) -> Wt[F,K] (row-major bf16)
__global__ void k_transpose_w(const float* __restrict__ W, bf16* __restrict__ Wt, int K, int F) {
  int i = blockIdx.x * blockDim.x + threadIdx.x;
  if (i < K * F) {
    int k = i / F, f = i - k * F;
    Wt[(size_t)f * K + k] = (bf16)W[i];
  }
}

// ---------------- bf16 WMMA GEMM, persistent waves, LDS-staged weights ---------
// C[M,F] = A[M,K] * Bt[F,K]^T.  The whole Bt (<=48KB) is async-copied into LDS
// once per block; each wave then loops over 16(M) x 64(N) output slabs, so the
// LDS weight copy is reused across many tiles.
// 16-bit A layout (matching B): lane<16 holds K 0..15, lane>=16 holds K 16..31.
// 16-bit B layout used here (same k order as A): lane&15 = column, per-lane 16 contiguous K values
// (lane half selects K 0..15 vs 16..31) -> contiguous rows of Bt.
template <int K, int F>
__global__ void k_gemm_lds(const bf16* __restrict__ A, const bf16* __restrict__ Bt,
                           const float* __restrict__ bias,
                           float* __restrict__ outF, bf16* __restrict__ outB,
                           int M, int relu) {
  __shared__ __align__(16) bf16 sB[F * K];

  // --- stage Bt into LDS via gfx1250 async-to-LDS path (16B per chunk) ---
  constexpr int CHUNKS = (F * K) / 8;          // 8 bf16 = 16 bytes per chunk
  for (int c = threadIdx.x; c < CHUNKS; c += blockDim.x) {
    unsigned loff = (unsigned)(uintptr_t)&sB[c * 8];   // LDS aperture: low 32b = offset
    async_g2lds_b128(loff, (const void*)(Bt + (size_t)c * 8));
  }
  wait_async0();
  __syncthreads();

  int lane = threadIdx.x & 31;
  int half = lane >> 4, l = lane & 15;
  constexpr int nGroups = F >> 6;              // F is a multiple of 64 for all calls
  int gridWaves = gridDim.x * (blockDim.x >> 5);
  int wave0 = blockIdx.x * (blockDim.x >> 5) + (threadIdx.x >> 5);
  int tiles = (M >> 4) * nGroups;

  for (int tile = wave0; tile < tiles; tile += gridWaves) {
    int mT = tile / nGroups;                   // nGroups is 1 or 2 -> shift
    int nG = tile - mT * nGroups;
    int m0 = mT << 4, n0 = nG << 6;

    const bf16* aP = A + (size_t)(m0 + l) * K + (half << 4);
    const bf16* bL = sB + (n0 + l) * K + (half << 4);

    v8f acc[4] = {};
#pragma unroll
    for (int k0 = 0; k0 < K; k0 += 32) {
      v16bf av = cat8(*(const v8bf*)(aP + k0), *(const v8bf*)(aP + k0 + 8));
      v16bf bv[4];
#pragma unroll
      for (int t = 0; t < 4; ++t) {            // distinct regs: all 8 ds_loads issue
        const bf16* bt = bL + t * (K << 4) + k0;
        bv[t] = cat8(*(const v8bf*)(bt), *(const v8bf*)(bt + 8));
      }
#pragma unroll
      for (int t = 0; t < 4; ++t)
        acc[t] = __builtin_amdgcn_wmma_f32_16x16x32_bf16(
            false, av, false, bv[t], (short)0, acc[t], false, false);
    }

    int row0 = m0 + (half << 3);               // C layout: lane half -> M 0..7 / 8..15
#pragma unroll
    for (int t = 0; t < 4; ++t) {
      int col = n0 + (t << 4) + l;
      float bb = bias ? bias[col] : 0.f;
#pragma unroll
      for (int v = 0; v < 8; ++v) {
        float val = acc[t][v] + bb;
        if (relu) val = fmaxf(val, 0.f);
        size_t o = (size_t)(row0 + v) * F + col;
        if (outF) outF[o] = val;
        if (outB) outB[o] = (bf16)val;
      }
    }
  }
}

// ---------------- edge scatter-add: agg[dst] += h[src] * norm ----------------
// One wave per edge: src/dst/norm become wave-uniform scalars; each lane moves
// F/32 features with a vector load + native f32 atomics.
template <int F>
__global__ void k_scatter(const int* __restrict__ src, const int* __restrict__ dst,
                          const float* __restrict__ enorm, const float* __restrict__ h,
                          float* __restrict__ agg, int E) {
  int wid = (blockIdx.x * blockDim.x + threadIdx.x) >> 5;
  int lane = threadIdx.x & 31;
  if (wid >= E) return;
  int s = __builtin_amdgcn_readfirstlane(src[wid]);
  int d = __builtin_amdgcn_readfirstlane(dst[wid]);
  float w = __uint_as_float(
      __builtin_amdgcn_readfirstlane(__float_as_uint(enorm[wid])));

  constexpr int V = F / 32;
  typedef float vecf __attribute__((ext_vector_type(V)));
  const float* hp = h + (size_t)s * F + lane * V;
  float* ap = agg + (size_t)d * F + lane * V;
  vecf hv = *(const vecf*)hp;
#pragma unroll
  for (int i = 0; i < V; ++i)
    __hip_atomic_fetch_add(ap + i, hv[i] * w, __ATOMIC_RELAXED, __HIP_MEMORY_SCOPE_AGENT);
}

// ---------------- agg init with self-loop term: agg = dinv^2 * h ----------------
__global__ void k_selfinit(const float* __restrict__ h, const float* __restrict__ dinv,
                           float* __restrict__ agg, int N, int logF) {
  long long i4 = (long long)blockIdx.x * blockDim.x + threadIdx.x;
  if (i4 >= (((long long)N << logF) >> 2)) return;
  int r = (int)((i4 << 2) >> logF);
  float di = dinv[r];
  float c = di * di;
  float4 hv = ((const float4*)h)[i4];
  ((float4*)agg)[i4] = make_float4(c * hv.x, c * hv.y, c * hv.z, c * hv.w);
}

// ---------------- epilogue: agg + b, ReLU, -> bf16 ----------------
__global__ void k_bias_relu_bf16(const float* __restrict__ agg, const float* __restrict__ bias,
                                 bf16* __restrict__ out, int N, int logF) {
  long long i4 = (long long)blockIdx.x * blockDim.x + threadIdx.x;
  if (i4 >= (((long long)N << logF) >> 2)) return;
  long long base = i4 << 2;
  int f = (int)(base & ((1 << logF) - 1));
  float4 a = ((const float4*)agg)[i4];
  out[base + 0] = (bf16)fmaxf(a.x + bias[f + 0], 0.f);
  out[base + 1] = (bf16)fmaxf(a.y + bias[f + 1], 0.f);
  out[base + 2] = (bf16)fmaxf(a.z + bias[f + 2], 0.f);
  out[base + 3] = (bf16)fmaxf(a.w + bias[f + 3], 0.f);
}

// ---------------- head: z[B,64] @ Wc3[64,2] + bc3, log_softmax ----------------
__global__ void k_head(const bf16* __restrict__ z, const float* __restrict__ Wc3,
                       const float* __restrict__ bc3, float* __restrict__ out, int B) {
  int b = blockIdx.x * blockDim.x + threadIdx.x;
  if (b >= B) return;
  const bf16* zr = z + (size_t)b * 64;
  float u0 = bc3[0], u1 = bc3[1];
#pragma unroll 4
  for (int k = 0; k < 64; ++k) {
    float zv = (float)zr[k];
    u0 += zv * Wc3[2 * k];
    u1 += zv * Wc3[2 * k + 1];
  }
  float m = fmaxf(u0, u1);
  float lse = m + logf(expf(u0 - m) + expf(u1 - m));
  out[2 * b + 0] = u0 - lse;
  out[2 * b + 1] = u1 - lse;
}

extern "C" void kernel_launch(void* const* d_in, const int* in_sizes, int n_in,
                              void* d_out, int out_size, void* d_ws, size_t ws_size,
                              hipStream_t stream) {
  const float* x   = (const float*)d_in[0];
  const int*   ei  = (const int*)d_in[1];
  const float* W1  = (const float*)d_in[2];  const float* b1  = (const float*)d_in[3];
  const float* W2  = (const float*)d_in[4];  const float* b2  = (const float*)d_in[5];
  const float* W3  = (const float*)d_in[6];  const float* b3  = (const float*)d_in[7];
  const float* Wc1 = (const float*)d_in[8];  const float* bc1 = (const float*)d_in[9];
  const float* Wc2 = (const float*)d_in[10]; const float* bc2 = (const float*)d_in[11];
  const float* Wc3 = (const float*)d_in[12]; const float* bc3 = (const float*)d_in[13];

  const int IN = 128, H = 128, P = 3;
  const int N = in_sizes[0] / IN;         // 150000
  const int E = in_sizes[1] / 2;          // 2400000
  const int B = N / P;                    // 50000
  const int* src = ei;
  const int* dst = ei + E;

  // workspace carve-out (256B aligned)
  char* ws = (char*)d_ws;
  size_t off = 0;
  auto take = [&](size_t bytes) {
    char* p = ws + off;
    off = (off + bytes + 255) & ~(size_t)255;
    return p;
  };
  float* dinv  = (float*)take((size_t)N * 4);
  float* enorm = (float*)take((size_t)E * 4);
  float* h     = (float*)take((size_t)N * H * 4);
  float* agg   = (float*)take((size_t)N * H * 4);
  bf16*  hbA   = (bf16*)take((size_t)N * H * 2);
  bf16*  hbB   = (bf16*)take((size_t)N * H * 2);
  bf16*  W1t   = (bf16*)take(128 * 128 * 2);
  bf16*  W2t   = (bf16*)take(128 * 128 * 2);
  bf16*  W3t   = (bf16*)take(128 * 64 * 2);
  bf16*  Wc1t  = (bf16*)take(192 * 128 * 2);
  bf16*  Wc2t  = (bf16*)take(128 * 64 * 2);
  (void)ws_size; (void)n_in; (void)out_size;

  auto cdiv = [](long long a, long long b) { return (int)((a + b - 1) / b); };
  auto gemm = [&](const bf16* A, const bf16* Bt, const float* bias,
                  float* oF, bf16* oB, int M, int K, int F, int relu) {
    int blocks = ((M >> 4) * (F >> 6) + 7) >> 3;   // 8 waves / block
    if (blocks > 1024) blocks = 1024;              // persistent waves loop over tiles
    if (K == 128 && F == 128)
      k_gemm_lds<128, 128><<<blocks, 256, 0, stream>>>(A, Bt, bias, oF, oB, M, relu);
    else if (K == 128 && F == 64)
      k_gemm_lds<128, 64><<<blocks, 256, 0, stream>>>(A, Bt, bias, oF, oB, M, relu);
    else
      k_gemm_lds<192, 128><<<blocks, 256, 0, stream>>>(A, Bt, bias, oF, oB, M, relu);
  };

  // ---- normalization prep (shared by all three layers) ----
  (void)hipMemsetAsync(dinv, 0, (size_t)N * 4, stream);
  k_degree<<<cdiv(E, 256), 256, 0, stream>>>(dst, dinv, E);
  k_dinv<<<cdiv(N, 256), 256, 0, stream>>>(dinv, N);
  k_edge_norm<<<cdiv(E, 256), 256, 0, stream>>>(src, dst, dinv, enorm, E);

  // ---- weight prep (bf16, transposed for WMMA B operand) ----
  k_transpose_w<<<cdiv(128 * 128, 256), 256, 0, stream>>>(W1, W1t, 128, 128);
  k_transpose_w<<<cdiv(128 * 128, 256), 256, 0, stream>>>(W2, W2t, 128, 128);
  k_transpose_w<<<cdiv(128 * 64, 256), 256, 0, stream>>>(W3, W3t, 128, 64);
  k_transpose_w<<<cdiv(192 * 128, 256), 256, 0, stream>>>(Wc1, Wc1t, 192, 128);
  k_transpose_w<<<cdiv(128 * 64, 256), 256, 0, stream>>>(Wc2, Wc2t, 128, 64);

  k_f32_to_bf16<<<cdiv((long long)N * IN, 256), 256, 0, stream>>>(x, hbA, N * IN);

  // ---- GCN layer 1: x -> hbB  (F=128) ----
  gemm(hbA, W1t, nullptr, h, nullptr, N, 128, 128, 0);
  k_selfinit<<<cdiv(((long long)N << 7) >> 2, 256), 256, 0, stream>>>(h, dinv, agg, N, 7);
  k_scatter<128><<<cdiv(E, 8), 256, 0, stream>>>(src, dst, enorm, h, agg, E);
  k_bias_relu_bf16<<<cdiv(((long long)N << 7) >> 2, 256), 256, 0, stream>>>(agg, b1, hbB, N, 7);

  // ---- GCN layer 2: hbB -> hbA  (F=128) ----
  gemm(hbB, W2t, nullptr, h, nullptr, N, 128, 128, 0);
  k_selfinit<<<cdiv(((long long)N << 7) >> 2, 256), 256, 0, stream>>>(h, dinv, agg, N, 7);
  k_scatter<128><<<cdiv(E, 8), 256, 0, stream>>>(src, dst, enorm, h, agg, E);
  k_bias_relu_bf16<<<cdiv(((long long)N << 7) >> 2, 256), 256, 0, stream>>>(agg, b2, hbA, N, 7);

  // ---- GCN layer 3: hbA -> hbB  (F=64); [N,64] bf16 == z[B,192] ----
  gemm(hbA, W3t, nullptr, h, nullptr, N, 128, 64, 0);
  k_selfinit<<<cdiv(((long long)N << 6) >> 2, 256), 256, 0, stream>>>(h, dinv, agg, N, 6);
  k_scatter<64><<<cdiv(E, 8), 256, 0, stream>>>(src, dst, enorm, h, agg, E);
  k_bias_relu_bf16<<<cdiv(((long long)N << 6) >> 2, 256), 256, 0, stream>>>(agg, b3, hbB, N, 6);

  // ---- classifier head (bias+ReLU fused into GEMM epilogue) ----
  gemm(hbB, Wc1t, bc1, nullptr, hbA, B, 192, 128, 1);   // z @ Wc1
  gemm(hbA, Wc2t, bc2, nullptr, hbB, B, 128, 64, 1);    // @ Wc2
  k_head<<<cdiv(B, 256), 256, 0, stream>>>(hbB, Wc3, bc3, (float*)d_out, B);
}
